// CausalLinearAttention_27015344292236
// MI455X (gfx1250) — compile-verified, matrix operand order checked by reading, not run on hardware
//
#include <hip/hip_runtime.h>
#include <hip/hip_bf16.h>

typedef __bf16 bf16_t;
typedef __attribute__((ext_vector_type(16))) __bf16 v16bf;
typedef __attribute__((ext_vector_type(8)))  __bf16 v8bf;
typedef __attribute__((ext_vector_type(8)))  float   v8f;

// ---------------------------------------------------------------------------
// WMMA helpers (CDNA5 gfx1250, wave32)
// ---------------------------------------------------------------------------
__device__ __forceinline__ v8f wmma_bf16(v16bf a, v16bf b, v8f c) {
    return __builtin_amdgcn_wmma_f32_16x16x32_bf16(
        /*neg_a=*/false, a, /*neg_b=*/false, b,
        /*c_mod=*/(short)0, c, /*reuse_a=*/false, /*reuse_b=*/false);
}

// Load a 16x32 bf16 fragment (A- or B-operand) from a K-contiguous matrix.
// CDNA5 layout: lane l holds row (row0 + l%16); lanes 0-15 carry
// K in [k0,k0+8) and [k0+16,k0+24); lanes 16-31 carry K in [k0+8,k0+16)
// and [k0+24,k0+32).
__device__ __forceinline__ v16bf load_frag(const bf16_t* __restrict__ base,
                                           int ld, int row0, int k0, int lane) {
    int r  = row0 + (lane & 15);
    int kb = k0 + ((lane >> 4) << 3);
    const bf16_t* p = base + r * ld + kb;
    v8bf lo = *(const v8bf*)(p);
    v8bf hi = *(const v8bf*)(p + 16);
    v16bf f;
#pragma unroll
    for (int i = 0; i < 8; ++i) { f[i] = lo[i]; f[i + 8] = hi[i]; }
    return f;
}

// ---------------------------------------------------------------------------
// CDNA5 async Global->LDS copy (ASYNCcnt-tracked DMA, no VGPR result).
// Per lane: LDS[ldsoff] = MEM[gaddr], 16 bytes.  Generic pointers to
// __shared__ carry the workgroup-relative LDS offset in their low 32 bits.
// ---------------------------------------------------------------------------
__device__ __forceinline__ unsigned lds_off32(const void* p) {
    return (unsigned)(unsigned long long)(uintptr_t)p;
}
__device__ __forceinline__ void async_copy_b128(unsigned ldsoff,
                                                unsigned long long gaddr) {
    asm volatile("global_load_async_to_lds_b128 %0, %1, off"
                 :: "v"(ldsoff), "v"(gaddr) : "memory");
}
__device__ __forceinline__ void wait_async0() {
    asm volatile("s_wait_asynccnt 0x0" ::: "memory");
}

// ---------------------------------------------------------------------------
// fp32 -> bf16 conversion
// ---------------------------------------------------------------------------
__global__ void f32_to_bf16_kernel(const float* __restrict__ in,
                                   bf16_t* __restrict__ out, long long n) {
    long long i = (long long)blockIdx.x * blockDim.x + threadIdx.x;
    long long stride = (long long)gridDim.x * blockDim.x;
    for (; i < n; i += stride) out[i] = (bf16_t)in[i];
}

// ---------------------------------------------------------------------------
// Double-buffered GEMM mainloop with async Global->LDS staging.
//   A : [M x 1024], Bm : [N x 1024], both bf16 K-contiguous
//   Block tile: 128(M) x 128(N), K-step 64, 8 waves in 4x2 (M x N) grid.
//   Wave computes 32x64 sub-tile: acc[nt*2 + mi], nt in [0,4), mi in [0,2).
// LDS: Ash/Bsh = 2 buffers x 128 rows x 64 bf16 (16 KB per buffer).
// Pipeline per iteration: issue async DMA for tile kt+1 into buffer nxt,
// run 16 WMMAs on buffer cur (DMA overlaps compute), s_wait_asynccnt 0,
// barrier, swap.
// ---------------------------------------------------------------------------
__device__ __forceinline__ void gemm_mainloop_128x128(
        const bf16_t* __restrict__ A, const bf16_t* __restrict__ Bm,
        int m0, int n0, bf16_t* Ash, bf16_t* Bsh, v8f acc[8], int tid) {
    const int NK = 16;                       // 1024 / 64 K-steps
    const int lane = tid & 31;
    const int wave = tid >> 5;
    const int wm = wave >> 1;                // 0..3 -> 32-row strip
    const int wn = wave & 1;                 // 0..1 -> 64-col strip
    const unsigned BUFB = 128 * 64 * 2;      // buffer stride in bytes

    // Per-thread cooperative-copy coordinates: 4 chunks of 16 bytes per tile.
    unsigned aoff[4], boff[4];
    unsigned long long ag[4], bg[4];
#pragma unroll
    for (int j = 0; j < 4; ++j) {
        int c = tid + 256 * j;               // chunk id in [0,1024)
        int row = c >> 3;                    // 8 chunks per 64-elem row
        int col = (c & 7) << 3;
        ag[j] = (unsigned long long)(uintptr_t)(A  + (m0 + row) * 1024 + col);
        bg[j] = (unsigned long long)(uintptr_t)(Bm + (n0 + row) * 1024 + col);
        aoff[j] = lds_off32(Ash + row * 64 + col);
        boff[j] = lds_off32(Bsh + row * 64 + col);
    }

    // ---- prologue: DMA tile 0 into buffer 0
#pragma unroll
    for (int j = 0; j < 4; ++j) {
        async_copy_b128(aoff[j], ag[j]);
        async_copy_b128(boff[j], bg[j]);
    }
    wait_async0();
    __syncthreads();

    int cur = 0;
    for (int kt = 0; kt < NK; ++kt) {
        // ---- issue DMA for next tile into the other buffer (overlaps WMMA)
        if (kt + 1 < NK) {
            unsigned long long kb = (unsigned long long)(kt + 1) * 64 * 2;
            unsigned bo = (unsigned)(cur ^ 1) * BUFB;
#pragma unroll
            for (int j = 0; j < 4; ++j) {
                async_copy_b128(aoff[j] + bo, ag[j] + kb);
                async_copy_b128(boff[j] + bo, bg[j] + kb);
            }
        }
        // ---- compute on current buffer
        const bf16_t* Ab = Ash + cur * (128 * 64);
        const bf16_t* Bb = Bsh + cur * (128 * 64);
#pragma unroll
        for (int kf = 0; kf < 2; ++kf) {
            v16bf af0 = load_frag(Ab, 64, wm * 32,      kf * 32, lane);
            v16bf af1 = load_frag(Ab, 64, wm * 32 + 16, kf * 32, lane);
#pragma unroll
            for (int nt = 0; nt < 4; ++nt) {
                v16bf bf = load_frag(Bb, 64, wn * 64 + nt * 16, kf * 32, lane);
                acc[nt * 2 + 0] = wmma_bf16(af0, bf, acc[nt * 2 + 0]);
                acc[nt * 2 + 1] = wmma_bf16(af1, bf, acc[nt * 2 + 1]);
            }
        }
        // ---- DMA for next buffer must land before anyone reads it
        wait_async0();
        __syncthreads();
        cur ^= 1;
    }
}

// ---------------------------------------------------------------------------
// Kernel A: qkv = x @ w_attn^T, apply phi to q/k, scatter to [B,H,T,D] bf16
// grid = (24, 64), block = 256
// ---------------------------------------------------------------------------
__global__ __launch_bounds__(256)
void qkv_gemm_kernel(const bf16_t* __restrict__ xb,
                     const bf16_t* __restrict__ wb,
                     bf16_t* __restrict__ qp,
                     bf16_t* __restrict__ kp,
                     bf16_t* __restrict__ vp) {
    __shared__ __align__(16) bf16_t Ash[2 * 128 * 64];
    __shared__ __align__(16) bf16_t Bsh[2 * 128 * 64];
    const int tid  = threadIdx.x;
    const int lane = tid & 31;
    const int wave = tid >> 5;
    const int m0 = blockIdx.y * 128;
    const int n0 = blockIdx.x * 128;

    v8f acc[8] = {};
    gemm_mainloop_128x128(xb, wb, m0, n0, Ash, Bsh, acc, tid);

    // Epilogue: phi + head scatter.  (A 128-wide N block never crosses a
    // 1024 boundary, so `part` is uniform per block.)
    const int wm = wave >> 1, wn = wave & 1;
    const int cn = lane & 15;
    const int mofs = (lane >> 4) << 3;
    const int part = (n0 >> 10);                 // 0=q, 1=k, 2=v
    bf16_t* dst = (part == 0) ? qp : (part == 1) ? kp : vp;
#pragma unroll
    for (int nt = 0; nt < 4; ++nt) {
        int col = n0 + wn * 64 + nt * 16 + cn;
        int e = col & 1023, h = e >> 6, d = e & 63;
#pragma unroll
        for (int mi = 0; mi < 2; ++mi)
#pragma unroll
            for (int v = 0; v < 8; ++v) {
                int m = m0 + wm * 32 + mi * 16 + v + mofs;
                float val = acc[nt * 2 + mi][v];
                if (part < 2) val = (val > 0.f) ? val + 1.f : __expf(val); // elu+1
                int b = m >> 11, t = m & 2047;
                dst[((long long)((b * 16 + h) * 2048 + t) << 6) + d] = (bf16_t)val;
            }
    }
}

// ---------------------------------------------------------------------------
// Kernel C: out = y @ w_proj^T, fp32 output.  grid = (8, 64), block = 256
// ---------------------------------------------------------------------------
__global__ __launch_bounds__(256)
void proj_gemm_kernel(const bf16_t* __restrict__ yb,
                      const bf16_t* __restrict__ wpb,
                      float* __restrict__ out) {
    __shared__ __align__(16) bf16_t Ash[2 * 128 * 64];
    __shared__ __align__(16) bf16_t Bsh[2 * 128 * 64];
    const int tid  = threadIdx.x;
    const int lane = tid & 31;
    const int wave = tid >> 5;
    const int m0 = blockIdx.y * 128;
    const int n0 = blockIdx.x * 128;

    v8f acc[8] = {};
    gemm_mainloop_128x128(yb, wpb, m0, n0, Ash, Bsh, acc, tid);

    const int wm = wave >> 1, wn = wave & 1;
    const int cn = lane & 15;
    const int mofs = (lane >> 4) << 3;
#pragma unroll
    for (int nt = 0; nt < 4; ++nt) {
        int col = n0 + wn * 64 + nt * 16 + cn;
#pragma unroll
        for (int mi = 0; mi < 2; ++mi)
#pragma unroll
            for (int v = 0; v < 8; ++v) {
                int m = m0 + wm * 32 + mi * 16 + v + mofs;
                out[(long long)m * 1024 + col] = acc[nt * 2 + mi][v];
            }
    }
}

// ---------------------------------------------------------------------------
// Kernel B: chunked causal linear attention scan, one workgroup per (b,h).
// ---------------------------------------------------------------------------
__global__ __launch_bounds__(256)
void attn_kernel(const bf16_t* __restrict__ qp,
                 const bf16_t* __restrict__ kp,
                 const bf16_t* __restrict__ vp,
                 bf16_t* __restrict__ yb) {
    __shared__ __align__(16) bf16_t qc[128 * 64];    // [c][d]
    __shared__ __align__(16) bf16_t kc[128 * 64];    // [key][d]
    __shared__ __align__(16) bf16_t kcT[64 * 128];   // [d][key]
    __shared__ __align__(16) bf16_t vcT[64 * 128];   // [e][key]
    __shared__ __align__(16) bf16_t sc[128 * 128];   // masked scores, bf16
    __shared__ __align__(16) float  Sf[64 * 64];     // S state fp32, [e][d]
    __shared__ __align__(16) bf16_t Sb[64 * 64];     // S state bf16, [e][d]
    __shared__ float zf[64];
    __shared__ float denb[128];

    const int tid  = threadIdx.x;
    const int lane = tid & 31;
    const int wave = tid >> 5;
    const int bh   = blockIdx.x;              // b*16 + h
    const int b    = bh >> 4;
    const int h    = bh & 15;
    const long long head_base = (long long)bh * 2048 * 64;

    for (int i = tid; i < 64 * 64; i += 256) { Sf[i] = 0.f; Sb[i] = (bf16_t)0.f; }
    if (tid < 64) zf[tid] = 0.f;
    __syncthreads();

    for (int nc = 0; nc < 16; ++nc) {
        const int t0 = nc * 128;
        const long long base = head_base + (long long)t0 * 64;

        // ---- stage q,k,v chunk into LDS (k,v also transposed) ----
        for (int i = tid; i < 128 * 8; i += 256) {
            int row = i >> 3, c8 = (i & 7) << 3;
            v8bf q8 = *(const v8bf*)(qp + base + row * 64 + c8);
            v8bf k8 = *(const v8bf*)(kp + base + row * 64 + c8);
            v8bf v8 = *(const v8bf*)(vp + base + row * 64 + c8);
            *(v8bf*)(qc + row * 64 + c8) = q8;
            *(v8bf*)(kc + row * 64 + c8) = k8;
#pragma unroll
            for (int j = 0; j < 8; ++j) {
                kcT[(c8 + j) * 128 + row] = k8[j];
                vcT[(c8 + j) * 128 + row] = v8[j];
            }
        }
        __syncthreads();

        // ---- scores = qc * kc^T  (wave w: rows [w*16, w*16+16), all 8 cols)
        {
            v8f sa[8] = {};
            v16bf a0 = load_frag(qc, 64, wave * 16, 0, lane);
            v16bf a1 = load_frag(qc, 64, wave * 16, 32, lane);
#pragma unroll
            for (int nt = 0; nt < 8; ++nt) {
                v16bf b0 = load_frag(kc, 64, nt * 16, 0, lane);
                sa[nt] = wmma_bf16(a0, b0, sa[nt]);
                v16bf b1 = load_frag(kc, 64, nt * 16, 32, lane);
                sa[nt] = wmma_bf16(a1, b1, sa[nt]);
            }
            const int cn = lane & 15;
            const int rofs = wave * 16 + ((lane >> 4) << 3);
#pragma unroll
            for (int nt = 0; nt < 8; ++nt)
#pragma unroll
                for (int v = 0; v < 8; ++v) {
                    int r = rofs + v;
                    int c = nt * 16 + cn;
                    float val = (c <= r) ? sa[nt][v] : 0.f;
                    sc[r * 128 + c] = (bf16_t)val;
                }
        }
        __syncthreads();

        // ---- den[r] = sum_c scores[r,c] + q[r]·z + eps  (threads 0..127)
        if (tid < 128) {
            float s = 0.f;
            for (int c = 0; c <= tid; ++c) s += (float)sc[tid * 128 + c];
            float qz = 0.f;
            for (int d = 0; d < 64; ++d) qz += (float)qc[tid * 64 + d] * zf[d];
            denb[tid] = s + qz + 1e-6f;
        }

        // ---- num = scores * v  +  q * S   (wave w: rows [w*16,+16), 4 e-tiles)
        v8f na[4] = {};
#pragma unroll
        for (int k0 = 0; k0 < 128; k0 += 32) {
            v16bf a = load_frag(sc, 128, wave * 16, k0, lane);
#pragma unroll
            for (int et = 0; et < 4; ++et) {
                v16bf bb = load_frag(vcT, 128, et * 16, k0, lane);
                na[et] = wmma_bf16(a, bb, na[et]);
            }
        }
#pragma unroll
        for (int k0 = 0; k0 < 64; k0 += 32) {
            v16bf a = load_frag(qc, 64, wave * 16, k0, lane);
#pragma unroll
            for (int et = 0; et < 4; ++et) {
                v16bf bb = load_frag(Sb, 64, et * 16, k0, lane);
                na[et] = wmma_bf16(a, bb, na[et]);
            }
        }
        __syncthreads();   // denb ready; all reads of old Sb done

        // ---- y = num / den, store bf16 to [B,T,E]
        {
            const int cn = lane & 15;
            const int rofs = wave * 16 + ((lane >> 4) << 3);
#pragma unroll
            for (int et = 0; et < 4; ++et)
#pragma unroll
                for (int v = 0; v < 8; ++v) {
                    int r = rofs + v;
                    int e = et * 16 + cn;
                    float y = na[et][v] / denb[r];
                    yb[(long long)(b * 2048 + t0 + r) * 1024 + h * 64 + e] = (bf16_t)y;
                }
        }

        // ---- S += k^T * v   (16 tiles of 16x16; wave w gets 2)
        {
            v8f ua[2] = {};
            const int dt = wave >> 1;
            const int e0 = (wave & 1) * 32;
#pragma unroll
            for (int k0 = 0; k0 < 128; k0 += 32) {
                v16bf a = load_frag(kcT, 128, dt * 16, k0, lane);
#pragma unroll
                for (int j = 0; j < 2; ++j) {
                    v16bf bb = load_frag(vcT, 128, e0 + j * 16, k0, lane);
                    ua[j] = wmma_bf16(a, bb, ua[j]);
                }
            }
            const int cn = lane & 15;
            const int dofs = dt * 16 + ((lane >> 4) << 3);
#pragma unroll
            for (int j = 0; j < 2; ++j)
#pragma unroll
                for (int v = 0; v < 8; ++v) {
                    int d = dofs + v;
                    int e = e0 + j * 16 + cn;
                    Sf[e * 64 + d] += ua[j][v];
                }
        }

        // ---- z += column sums of k
        if (tid < 64) {
            float s = 0.f;
            for (int kk = 0; kk < 128; ++kk) s += (float)kcT[tid * 128 + kk];
            zf[tid] += s;
        }
        __syncthreads();   // Sf, zf fully updated

        // ---- refresh bf16 mirror of S
        for (int i = tid; i < 64 * 64; i += 256) Sb[i] = (bf16_t)Sf[i];
        __syncthreads();
    }
}

// ---------------------------------------------------------------------------
// Launcher
// ---------------------------------------------------------------------------
extern "C" void kernel_launch(void* const* d_in, const int* in_sizes, int n_in,
                              void* d_out, int out_size, void* d_ws, size_t ws_size,
                              hipStream_t stream) {
    const float* x      = (const float*)d_in[0];   // [4,2048,1024]
    const float* w_attn = (const float*)d_in[1];   // [3072,1024]
    const float* w_proj = (const float*)d_in[2];   // [1024,1024]
    float* out = (float*)d_out;                    // [4,2048,1024]

    const long long NX = 8388608;   // 4*2048*1024
    const long long NWA = 3145728;  // 3072*1024
    const long long NWP = 1048576;  // 1024*1024

    bf16_t* p   = (bf16_t*)d_ws;
    bf16_t* xb  = p; p += NX;
    bf16_t* wab = p; p += NWA;
    bf16_t* wpb = p; p += NWP;
    bf16_t* qp  = p; p += NX;
    bf16_t* kp  = p; p += NX;
    bf16_t* vp  = p; p += NX;
    bf16_t* yb  = p; p += NX;

    f32_to_bf16_kernel<<<2048, 256, 0, stream>>>(x, xb, NX);
    f32_to_bf16_kernel<<<1024, 256, 0, stream>>>(w_attn, wab, NWA);
    f32_to_bf16_kernel<<<512,  256, 0, stream>>>(w_proj, wpb, NWP);

    qkv_gemm_kernel<<<dim3(24, 64), 256, 0, stream>>>(xb, wab, qp, kp, vp);
    attn_kernel<<<64, 256, 0, stream>>>(qp, kp, vp, yb);
    proj_gemm_kernel<<<dim3(8, 64), 256, 0, stream>>>(yb, wpb, out);
}
